// AttentionLayer_11708080849371
// MI455X (gfx1250) — matrix operand order repaired and checked by reading, not compiled, not run on hardware
//
#include <hip/hip_runtime.h>
#include <math.h>

typedef float v8f __attribute__((ext_vector_type(8)));
typedef __bf16 bf16_t;
typedef __bf16 v4bf __attribute__((ext_vector_type(4)));
typedef __bf16 v8bf __attribute__((ext_vector_type(8)));
typedef __bf16 v16bf __attribute__((ext_vector_type(16)));

#define BB 64
#define TT 1024
#define DD 512
#define UU 1024
#define APITCH 520  // bf16 elements per LDS row: 512 + 8 pad (1040 B, 16B-aligned)
#define MROWS 32    // T-rows per block: two 16-row WMMA tiles share B fragments

// ---------------------------------------------------------------------------
// Kernel 0: pack W1 (fp32 [D,U]) into split-bf16 hi/lo arrays in B-fragment
// order: dst[((kt*2 + hi)*U + n)*16 + j] = W1[(kt*32 + hi*8 + (j&7) + (j>>3)*16)*U + n]
// so each WMMA lane reads its 16 K-elements as one 32B contiguous chunk.
// ---------------------------------------------------------------------------
__global__ void __launch_bounds__(256) pack_W1_kernel(
    const float* __restrict__ W1, bf16_t* __restrict__ Whi,
    bf16_t* __restrict__ Wlo) {
  const int g = blockIdx.x * 256 + threadIdx.x;  // 0 .. D*U-1
  const int j = g & 15;
  const int n = (g >> 4) & (UU - 1);
  const int kh = g >> 14;           // kt*2 + hi
  const int d = (kh >> 1) * 32 + (kh & 1) * 8 + (j & 7) + (j >> 3) * 16;
  const float v = W1[(size_t)d * UU + n];
  const __bf16 h = (__bf16)v;
  Whi[g] = h;
  Wlo[g] = (__bf16)(v - (float)h);
}

// ---------------------------------------------------------------------------
// Kernel 1: ph[b,u] = hidden[b,:]·W2[:,u] + W2_b[u] + W1_b[u]
// (V_b dropped: softmax is shift-invariant and logits aren't an output)
// ---------------------------------------------------------------------------
__global__ void __launch_bounds__(256) proj_h_kernel(
    const float* __restrict__ hidden, const float* __restrict__ W2,
    const float* __restrict__ W2b, const float* __restrict__ W1b,
    float* __restrict__ ph) {
  __shared__ float h[DD];
  const int b = blockIdx.x, tid = threadIdx.x;
  h[tid]       = hidden[b * DD + tid];
  h[tid + 256] = hidden[b * DD + tid + 256];
  __syncthreads();
  float acc[4];
#pragma unroll
  for (int i = 0; i < 4; ++i)
    acc[i] = W2b[tid + i * 256] + W1b[tid + i * 256];
  for (int d = 0; d < DD; ++d) {
    const float hv = h[d];
    const float* w = W2 + (size_t)d * UU + tid;
#pragma unroll
    for (int i = 0; i < 4; ++i) acc[i] = fmaf(hv, w[i * 256], acc[i]);
  }
#pragma unroll
  for (int i = 0; i < 4; ++i) ph[(size_t)b * UU + tid + i * 256] = acc[i];
}

// ---------------------------------------------------------------------------
// Kernel 2: fused logits[b,t] = sum_u tanh((feat@W1)[t,u] + ph[b,u]) * V[u]
// One block = 32 rows of T (2 M-tiles). 8 waves x 8 N-tiles of 16 U-cols.
// Split-bf16 GEMM: per K=32 slab and M-tile, 3x v_wmma_f32_16x16x32_bf16
// (hi*hi + lo*hi + hi*lo), fp32 accumulate -> ~1.5e-5 relative error.
// Two M-tiles share each B fragment: halves W1/L2 traffic per FLOP.
// ---------------------------------------------------------------------------
__global__ void __launch_bounds__(256) score_logits_kernel(
    const float* __restrict__ feat, const bf16_t* __restrict__ W1hi,
    const bf16_t* __restrict__ W1lo, const float* __restrict__ ph,
    const float* __restrict__ Vw, float* __restrict__ logits) {
  __shared__ bf16_t Ahi[MROWS * APITCH];
  __shared__ bf16_t Alo[MROWS * APITCH];
  __shared__ float lpart[MROWS];

  const int tid = threadIdx.x;
  const int blk = blockIdx.x;
  const int b   = blk / (TT / MROWS);
  const int t0  = (blk % (TT / MROWS)) * MROWS;

  if (tid < MROWS) lpart[tid] = 0.0f;

  // stage 32x512 fp32 feature tile as split-bf16 hi/lo in LDS
  const float* src = feat + (size_t)b * TT * DD + (size_t)t0 * DD;
#pragma unroll
  for (int i = 0; i < 16; ++i) {
    const int flat = (tid + i * 256) * 4;  // 0..16380
    const int r = flat >> 9;
    const int c = flat & 511;
    const float4 v = *(const float4*)(src + (size_t)r * DD + c);
    const float fs[4] = {v.x, v.y, v.z, v.w};
    v4bf h4, l4;
#pragma unroll
    for (int e = 0; e < 4; ++e) {
      const __bf16 hb = (__bf16)fs[e];
      h4[e] = hb;
      l4[e] = (__bf16)(fs[e] - (float)hb);
    }
    *(v4bf*)(&Ahi[r * APITCH + c]) = h4;
    *(v4bf*)(&Alo[r * APITCH + c]) = l4;
  }
  __syncthreads();

  const int lane = tid & 31;
  const int wave = tid >> 5;
  const int hi   = lane >> 4;          // half-wave
  const int ln   = lane & 15;          // N within tile (and A source row)
  const int colbase = wave * 128 + ln;

  // A frags: K = k0+8*hi+{0..7} and k0+16+8*hi+{0..7}  (ISA 16-bit A 16x32)
  const bf16_t* arow_h0 = &Ahi[ln * APITCH + 8 * hi];
  const bf16_t* arow_l0 = &Alo[ln * APITCH + 8 * hi];
  const bf16_t* arow_h1 = &Ahi[(16 + ln) * APITCH + 8 * hi];
  const bf16_t* arow_l1 = &Alo[(16 + ln) * APITCH + 8 * hi];
  // B frags: packed contiguous 16 K-elements per (kt,hi,col)
  const bf16_t* pBh = W1hi + ((size_t)hi * UU + colbase) * 16;
  const bf16_t* pBl = W1lo + ((size_t)hi * UU + colbase) * 16;

  v8f acc0[8] = {};
  v8f acc1[8] = {};

  for (int kt = 0; kt < DD / 32; ++kt) {
    const int k0 = kt * 32;
    const v16bf a_h0 = __builtin_shufflevector(
        *(const v8bf*)(arow_h0 + k0), *(const v8bf*)(arow_h0 + k0 + 16), 0, 1,
        2, 3, 4, 5, 6, 7, 8, 9, 10, 11, 12, 13, 14, 15);
    const v16bf a_l0 = __builtin_shufflevector(
        *(const v8bf*)(arow_l0 + k0), *(const v8bf*)(arow_l0 + k0 + 16), 0, 1,
        2, 3, 4, 5, 6, 7, 8, 9, 10, 11, 12, 13, 14, 15);
    const v16bf a_h1 = __builtin_shufflevector(
        *(const v8bf*)(arow_h1 + k0), *(const v8bf*)(arow_h1 + k0 + 16), 0, 1,
        2, 3, 4, 5, 6, 7, 8, 9, 10, 11, 12, 13, 14, 15);
    const v16bf a_l1 = __builtin_shufflevector(
        *(const v8bf*)(arow_l1 + k0), *(const v8bf*)(arow_l1 + k0 + 16), 0, 1,
        2, 3, 4, 5, 6, 7, 8, 9, 10, 11, 12, 13, 14, 15);
    const bf16_t* bh = pBh + (size_t)kt * (2 * UU * 16);
    const bf16_t* bl = pBl + (size_t)kt * (2 * UU * 16);
#pragma unroll
    for (int j = 0; j < 8; ++j) {
      const v16bf b_hi = *(const v16bf*)(bh + j * 256);
      const v16bf b_lo = *(const v16bf*)(bl + j * 256);
      acc0[j] = __builtin_amdgcn_wmma_f32_16x16x32_bf16(
          false, a_h0, false, b_hi, (short)0, acc0[j], false, false);
      acc1[j] = __builtin_amdgcn_wmma_f32_16x16x32_bf16(
          false, a_h1, false, b_hi, (short)0, acc1[j], false, false);
      acc0[j] = __builtin_amdgcn_wmma_f32_16x16x32_bf16(
          false, a_l0, false, b_hi, (short)0, acc0[j], false, false);
      acc1[j] = __builtin_amdgcn_wmma_f32_16x16x32_bf16(
          false, a_l1, false, b_hi, (short)0, acc1[j], false, false);
      acc0[j] = __builtin_amdgcn_wmma_f32_16x16x32_bf16(
          false, a_h0, false, b_lo, (short)0, acc0[j], false, false);
      acc1[j] = __builtin_amdgcn_wmma_f32_16x16x32_bf16(
          false, a_h1, false, b_lo, (short)0, acc1[j], false, false);
    }
  }

  // epilogue: tanh(+bias) * V, reduce over U for both M-tiles
  float part0[8], part1[8];
#pragma unroll
  for (int r = 0; r < 8; ++r) { part0[r] = 0.0f; part1[r] = 0.0f; }
  const float* phb = ph + (size_t)b * UU;
#pragma unroll
  for (int j = 0; j < 8; ++j) {
    const int n = colbase + j * 16;
    const float bias = phb[n];
    const float vw = Vw[n];
#pragma unroll
    for (int r = 0; r < 8; ++r) {
      part0[r] = fmaf(tanhf(acc0[j][r] + bias), vw, part0[r]);
      part1[r] = fmaf(tanhf(acc1[j][r] + bias), vw, part1[r]);
    }
  }
#pragma unroll
  for (int m = 1; m < 16; m <<= 1) {
#pragma unroll
    for (int r = 0; r < 8; ++r) {
      part0[r] += __shfl_xor(part0[r], m, 32);
      part1[r] += __shfl_xor(part1[r], m, 32);
    }
  }
  if (ln == 0) {
#pragma unroll
    for (int r = 0; r < 8; ++r) {
      atomicAdd(&lpart[r + 8 * hi], part0[r]);
      atomicAdd(&lpart[16 + r + 8 * hi], part1[r]);
    }
  }
  __syncthreads();
  if (tid < MROWS) logits[(size_t)b * TT + t0 + tid] = lpart[tid];
}

// ---------------------------------------------------------------------------
// Kernel 3: softmax over T per b; writes attention weights, zeroes context
// ---------------------------------------------------------------------------
__global__ void __launch_bounds__(256) softmax_kernel(
    const float* __restrict__ logits, float* __restrict__ attw,
    float* __restrict__ ctx) {
  __shared__ float red[256];
  const int b = blockIdx.x, tid = threadIdx.x;
  const float4 v = *(const float4*)(logits + (size_t)b * TT + tid * 4);

  float m = fmaxf(fmaxf(v.x, v.y), fmaxf(v.z, v.w));
  red[tid] = m;
  __syncthreads();
  for (int s = 128; s > 0; s >>= 1) {
    if (tid < s) red[tid] = fmaxf(red[tid], red[tid + s]);
    __syncthreads();
  }
  const float gm = red[0];
  __syncthreads();

  const float e0 = expf(v.x - gm), e1 = expf(v.y - gm);
  const float e2 = expf(v.z - gm), e3 = expf(v.w - gm);
  red[tid] = e0 + e1 + e2 + e3;
  __syncthreads();
  for (int s = 128; s > 0; s >>= 1) {
    if (tid < s) red[tid] += red[tid + s];
    __syncthreads();
  }
  const float inv = 1.0f / red[0];

  float4 w;
  w.x = e0 * inv; w.y = e1 * inv; w.z = e2 * inv; w.w = e3 * inv;
  *(float4*)(attw + (size_t)b * TT + tid * 4) = w;

  ctx[(size_t)b * DD + tid] = 0.0f;
  ctx[(size_t)b * DD + tid + 256] = 0.0f;
}

// ---------------------------------------------------------------------------
// Kernel 4: context[b,d] = sum_t w[b,t] * feat[b,t,d]; 8 T-chunks per b
// ---------------------------------------------------------------------------
__global__ void __launch_bounds__(256) context_kernel(
    const float* __restrict__ feat, const float* __restrict__ attw,
    float* __restrict__ ctx) {
  __shared__ float wl[128];
  const int blk = blockIdx.x;
  const int b = blk >> 3;
  const int t0 = (blk & 7) * 128;
  const int tid = threadIdx.x;
  if (tid < 128) wl[tid] = attw[(size_t)b * TT + t0 + tid];
  __syncthreads();
  const float* fb = feat + (size_t)b * TT * DD + (size_t)t0 * DD;
  float a0 = 0.0f, a1 = 0.0f;
  for (int t = 0; t < 128; ++t) {
    const float* row = fb + (size_t)t * DD;
    const float wv = wl[t];
    a0 = fmaf(wv, row[tid], a0);
    a1 = fmaf(wv, row[tid + 256], a1);
  }
  unsafeAtomicAdd(&ctx[(size_t)b * DD + tid], a0);
  unsafeAtomicAdd(&ctx[(size_t)b * DD + tid + 256], a1);
}

// ---------------------------------------------------------------------------
extern "C" void kernel_launch(void* const* d_in, const int* in_sizes, int n_in,
                              void* d_out, int out_size, void* d_ws,
                              size_t ws_size, hipStream_t stream) {
  (void)in_sizes; (void)n_in; (void)out_size; (void)ws_size;
  const float* feat   = (const float*)d_in[0];  // [B,T,D]
  const float* hidden = (const float*)d_in[1];  // [B,D]
  const float* W1w    = (const float*)d_in[2];  // [D,U]
  const float* W1b    = (const float*)d_in[3];  // [U]
  const float* W2w    = (const float*)d_in[4];  // [D,U]
  const float* W2b    = (const float*)d_in[5];  // [U]
  const float* Vw     = (const float*)d_in[6];  // [U,1]
  // d_in[7] = V_b: unused (softmax shift-invariance)

  float* ctx  = (float*)d_out;            // [B,D]
  float* attw = (float*)d_out + BB * DD;  // [B,T]

  // workspace layout (all offsets 256B-aligned)
  float*  ph     = (float*)d_ws;                              // B*U fp32
  float*  logits = (float*)d_ws + BB * UU;                    // B*T fp32
  bf16_t* W1hi   = (bf16_t*)((char*)d_ws + 512 * 1024);       // D*U bf16
  bf16_t* W1lo   = W1hi + (size_t)DD * UU;                    // D*U bf16

  pack_W1_kernel<<<DD * UU / 256, 256, 0, stream>>>(W1w, W1hi, W1lo);
  proj_h_kernel<<<BB, 256, 0, stream>>>(hidden, W2w, W2b, W1b, ph);
  score_logits_kernel<<<BB * TT / MROWS, 256, 0, stream>>>(feat, W1hi, W1lo,
                                                           ph, Vw, logits);
  softmax_kernel<<<BB, 256, 0, stream>>>(logits, attw, ctx);
  context_kernel<<<BB * 8, 256, 0, stream>>>(feat, attw, ctx);
}
